// DTFDynamicLayer_48507360641339
// MI455X (gfx1250) — matrix operand order repaired and checked by reading, not compiled, not run on hardware
//
#include <hip/hip_runtime.h>

// ---------------------------------------------------------------------------
// Problem constants (from the reference: B,S,D,H,KV,HD,DFF,k are fixed)
// ---------------------------------------------------------------------------
static constexpr int BB   = 4;
static constexpr int SS   = 4096;
static constexpr int DD   = 2048;
static constexpr int HH   = 16;
static constexpr int KVH  = 4;
static constexpr int HDIM = 128;
static constexpr int DFF  = 8192;
static constexpr int KSEL = 512;
static constexpr int TT   = BB * KSEL;   // 2048 selected tokens
static constexpr float EPS = 1e-6f;

typedef __bf16 bf16;
typedef __attribute__((ext_vector_type(16))) __bf16 bf16x16;
typedef __attribute__((ext_vector_type(8)))  __bf16 bf16x8;
typedef __attribute__((ext_vector_type(8)))  float  f32x8;
typedef __attribute__((ext_vector_type(4)))  unsigned int uint32x4;
typedef __attribute__((ext_vector_type(8)))  int  int32x8;
typedef __attribute__((ext_vector_type(4)))  int  int32x4;

union Frag { bf16x16 v; bf16x8 h[2]; };

__device__ __forceinline__ f32x8 wmma_bf16(bf16x16 a, bf16x16 b, f32x8 c) {
  return __builtin_amdgcn_wmma_f32_16x16x32_bf16(false, a, false, b, (short)0, c,
                                                 false, false);
}

__device__ __forceinline__ f32x8 fzero8() {
  f32x8 v;
#pragma unroll
  for (int i = 0; i < 8; ++i) v[i] = 0.0f;
  return v;
}

// CDNA5 async global->LDS copy (16B per lane), tracked by ASYNCcnt.
// vdst = per-lane LDS byte offset (flat shared addr low 32 bits), vaddr = global.
__device__ __forceinline__ void async_copy_b128(void* lds_ptr, const void* gaddr) {
  unsigned lds_off = (unsigned)(uintptr_t)lds_ptr;  // LDS aperture: offset in [31:0]
  asm volatile("global_load_async_to_lds_b128 %0, %1, off"
               :
               : "v"(lds_off), "v"((unsigned long long)(uintptr_t)gaddr)
               : "memory");
}
__device__ __forceinline__ void wait_async_le2() {
  asm volatile("s_wait_asynccnt 0x2" ::: "memory");
}
__device__ __forceinline__ void wait_async_le0() {
  asm volatile("s_wait_asynccnt 0x0" ::: "memory");
}

// CDNA5 Tensor Data Mover: DMA a 2-D bf16 tile [rows=128 x cols=64] from global
// (row stride = strideElems) into LDS with hardware row padding:
// pad_interval=4 -> every 32 DWORDs (64 bf16), pad_amount=7 -> 8 DWORDs (16 bf16)
// => LDS row stride of 80 bf16, matching As[][80]. Issued once per wave (EXEC
// ignored); tracked by TENSORcnt.
__device__ __forceinline__ void tdm_load_tile_a(const void* gaddr, void* lds_ptr,
                                                unsigned strideElems) {
  unsigned long long ga = (unsigned long long)(uintptr_t)gaddr;
  uint32x4 g0;
  g0[0] = 1u;                                       // count=1 valid descriptor
  g0[1] = (unsigned)(uintptr_t)lds_ptr;             // lds_addr (bytes)
  g0[2] = (unsigned)(ga & 0xFFFFFFFFu);             // global_addr[31:0]
  g0[3] = (unsigned)((ga >> 32) & 0x01FFFFFFu) | (2u << 30);  // [56:32] + type=2
  int32x8 g1;
  g1[0] = (1 << 16) | (1 << 20) | (4 << 22) | (7 << 25);  // 2B, pad 32DW->+8DW
  const unsigned td0 = 1u << 24, td1 = 1u << 24;          // big tensor dims (no OOB)
  g1[1] = (int)((td0 & 0xFFFFu) << 16);                   // tensor_dim0[15:0]
  g1[2] = (int)(((td0 >> 16) & 0xFFFFu) | ((td1 & 0xFFFFu) << 16));
  g1[3] = (int)(((td1 >> 16) & 0xFFFFu) | (64u << 16));   // tile_dim0 = 64
  g1[4] = (int)128u;                                      // tile_dim1=128, tile_dim2=0
  g1[5] = (int)strideElems;                               // tensor_dim0_stride lo32
  g1[6] = 0;                                              // stride hi16 | dim1_stride lo
  g1[7] = 0;
  int32x4 gz;
  gz[0] = 0; gz[1] = 0; gz[2] = 0; gz[3] = 0;
#if defined(__clang_major__) && (__clang_major__ >= 23)
  int32x8 gz8;
#pragma unroll
  for (int i = 0; i < 8; ++i) gz8[i] = 0;
  __builtin_amdgcn_tensor_load_to_lds(g0, g1, gz, gz, gz8, 0);
#else
  __builtin_amdgcn_tensor_load_to_lds(g0, g1, gz, gz, 0);
#endif
}

// ---------------------------------------------------------------------------
// 1) Router scores: scores[b,s] = original . w_router + 0.5*mean((post-prior)^2)
// ---------------------------------------------------------------------------
__global__ __launch_bounds__(256)
void router_kernel(const float* __restrict__ orig, const float* __restrict__ post,
                   const float* __restrict__ pri, const float* __restrict__ wr,
                   float* __restrict__ scores) {
  __shared__ float r1[256], r2[256];
  const size_t tkn = blockIdx.x;
  const float* o  = orig + tkn * DD;
  const float* ps = post + tkn * DD;
  const float* pr = pri  + tkn * DD;
  float dot = 0.f, sq = 0.f;
  for (int d = threadIdx.x; d < DD; d += 256) {
    dot += o[d] * wr[d];
    float df = ps[d] - pr[d];
    sq += df * df;
  }
  r1[threadIdx.x] = dot; r2[threadIdx.x] = sq;
  __syncthreads();
  for (int st = 128; st > 0; st >>= 1) {
    if (threadIdx.x < st) {
      r1[threadIdx.x] += r1[threadIdx.x + st];
      r2[threadIdx.x] += r2[threadIdx.x + st];
    }
    __syncthreads();
  }
  if (threadIdx.x == 0) scores[tkn] = r1[0] + 0.5f * r2[0] / (float)DD;
}

// ---------------------------------------------------------------------------
// 2) Per-batch top-k via bitonic sort of packed (score,idx) keys, then sort
//    the selected indices ascending (matches sort(top_k(...)[1])).
// ---------------------------------------------------------------------------
__global__ __launch_bounds__(1024)
void topk_kernel(const float* __restrict__ scores, int* __restrict__ idxOut,
                 float* __restrict__ gateOut) {
  __shared__ unsigned long long keys[SS];
  __shared__ int sidx[KSEL];
  const int b = blockIdx.x, tid = threadIdx.x;
  for (int i = tid; i < SS; i += 1024) {
    unsigned u = __float_as_uint(scores[b * SS + i]);
    u = (u & 0x80000000u) ? ~u : (u | 0x80000000u);  // order-preserving map
    keys[i] = ((unsigned long long)(~u) << 32) | (unsigned)i; // asc = desc score
  }
  __syncthreads();
  for (int sz = 2; sz <= SS; sz <<= 1)
    for (int st = sz >> 1; st > 0; st >>= 1) {
      for (int i = tid; i < SS; i += 1024) {
        int j = i ^ st;
        if (j > i) {
          bool up = ((i & sz) == 0);
          unsigned long long a = keys[i], c = keys[j];
          if ((a > c) == up) { keys[i] = c; keys[j] = a; }
        }
      }
      __syncthreads();
    }
  for (int i = tid; i < KSEL; i += 1024)
    sidx[i] = (int)(keys[i] & 0xFFFFFFFFull);
  __syncthreads();
  for (int sz = 2; sz <= KSEL; sz <<= 1)
    for (int st = sz >> 1; st > 0; st >>= 1) {
      for (int i = tid; i < KSEL; i += 1024) {
        int j = i ^ st;
        if (j > i) {
          bool up = ((i & sz) == 0);
          int a = sidx[i], c = sidx[j];
          if ((a > c) == up) { sidx[i] = c; sidx[j] = a; }
        }
      }
      __syncthreads();
    }
  for (int i = tid; i < KSEL; i += 1024) {
    int s = sidx[i];
    idxOut[b * KSEL + i] = s;
    float sc = scores[b * SS + s];
    gateOut[b * KSEL + i] = 1.0f / (1.0f + __expf(-sc));
  }
}

// ---------------------------------------------------------------------------
// 3) Gather selected tokens + RMSNorm1 -> bf16 activations
// ---------------------------------------------------------------------------
__global__ __launch_bounds__(256)
void gather_rms_kernel(const float* __restrict__ hidden, const int* __restrict__ idxb,
                       const int* __restrict__ posids, const float* __restrict__ ln1,
                       float* __restrict__ sel, bf16* __restrict__ h1,
                       int* __restrict__ poss) {
  __shared__ float red[256];
  __shared__ float rs_sh;
  const int t = blockIdx.x, b = t / KSEL;
  const int s = idxb[t];
  const float* x = hidden + ((size_t)b * SS + s) * DD;
  float sq = 0.f;
  for (int d = threadIdx.x; d < DD; d += 256) { float v = x[d]; sq += v * v; }
  red[threadIdx.x] = sq; __syncthreads();
  for (int st = 128; st > 0; st >>= 1) {
    if (threadIdx.x < st) red[threadIdx.x] += red[threadIdx.x + st];
    __syncthreads();
  }
  if (threadIdx.x == 0) {
    rs_sh = rsqrtf(red[0] / (float)DD + EPS);
    poss[t] = posids[b * SS + s];
  }
  __syncthreads();
  const float rs = rs_sh;
  for (int d = threadIdx.x; d < DD; d += 256) {
    float v = x[d];
    sel[(size_t)t * DD + d] = v;
    h1[(size_t)t * DD + d]  = (bf16)(v * rs * ln1[d]);
  }
}

// RMSNorm2 (no gather)
__global__ __launch_bounds__(256)
void rms2_kernel(const float* __restrict__ x1, const float* __restrict__ ln2,
                 bf16* __restrict__ h2) {
  __shared__ float red[256];
  __shared__ float rs_sh;
  const int t = blockIdx.x;
  const float* x = x1 + (size_t)t * DD;
  float sq = 0.f;
  for (int d = threadIdx.x; d < DD; d += 256) { float v = x[d]; sq += v * v; }
  red[threadIdx.x] = sq; __syncthreads();
  for (int st = 128; st > 0; st >>= 1) {
    if (threadIdx.x < st) red[threadIdx.x] += red[threadIdx.x + st];
    __syncthreads();
  }
  if (threadIdx.x == 0) rs_sh = rsqrtf(red[0] / (float)DD + EPS);
  __syncthreads();
  const float rs = rs_sh;
  for (int d = threadIdx.x; d < DD; d += 256)
    h2[(size_t)t * DD + d] = (bf16)(x[d] * rs * ln2[d]);
}

// ---------------------------------------------------------------------------
// 4) Weight fp32 [K,N] -> bf16 transposed [N,K] (B operand becomes N-major so
//    WMMA B-fragments are contiguous per-lane K-chunks)
// ---------------------------------------------------------------------------
__global__ void wconvT_kernel(const float* __restrict__ W, bf16* __restrict__ Wt,
                              int K, int N) {
  size_t i = (size_t)blockIdx.x * 256 + threadIdx.x;
  if (i >= (size_t)K * N) return;
  size_t n = i / (size_t)K, kk = i % (size_t)K;
  Wt[i] = (bf16)W[kk * (size_t)N + n];
}

// ---------------------------------------------------------------------------
// 5) WMMA GEMM:  D[M,N] = A[M,K](bf16,row) x Bt[N,K](bf16,row)  (+ residual)
//    block = 256 thr (8 waves), tile 128x128, BK=64, wave tile 64x32.
//    A tile staged by the Tensor Data Mover (TENSORcnt, HW row padding);
//    B tile staged by GLOBAL_LOAD_ASYNC_TO_LDS_B128 (ASYNCcnt); double-buffered.
// ---------------------------------------------------------------------------
__global__ __launch_bounds__(256)
void gemm_bf16_kernel(const bf16* __restrict__ A, const bf16* __restrict__ Bt,
                      float* __restrict__ Dm, const float* __restrict__ resid,
                      int M, int N, int K) {
  __shared__ __align__(16) bf16 As[2][128][80];   // stride 80 -> 160B rows
  __shared__ __align__(16) bf16 Bs[2][128][80];
  const int tid  = threadIdx.x;
  const int lane = tid & 31, l = lane & 15, hh = lane >> 4;
  const int w = tid >> 5, wm = w >> 2, wn = w & 3;
  const bool wv0 = (w == 0);                       // wave-uniform
  const int m0 = blockIdx.y * 128, n0 = blockIdx.x * 128;

  // per-thread B staging coordinates: 512 chunks of 8 bf16, 2 per thread
  const int rowA = tid >> 3,          kcA = (tid & 7) << 3;
  const int rowB = (tid + 256) >> 3,  kcB = ((tid + 256) & 7) << 3;

  // stage one BK=64 tile: A via TDM (wave 0), Bt via async copies (all threads)
  auto stage = [&](int buf, int k0) {
    if (wv0)
      tdm_load_tile_a(A + (size_t)m0 * K + k0, &As[buf][0][0], (unsigned)K);
    async_copy_b128(&Bs[buf][rowA][kcA], Bt + (size_t)(n0 + rowA) * K + k0 + kcA);
    async_copy_b128(&Bs[buf][rowB][kcB], Bt + (size_t)(n0 + rowB) * K + k0 + kcB);
  };

  f32x8 acc[4][2];
#pragma unroll
  for (int mi = 0; mi < 4; ++mi)
#pragma unroll
    for (int ni = 0; ni < 2; ++ni) acc[mi][ni] = fzero8();

  stage(0, 0);  // preload first tile

  for (int k0 = 0; k0 < K; k0 += 64) {
    const int cur = (k0 >> 6) & 1;
    const bool more = (k0 + 64) < K;
    if (more) stage(cur ^ 1, k0 + 64);   // prefetch next tile
    // loads complete in order: current tile's done once <= prefetched remain
    if (more) {
      wait_async_le2();
      if (wv0) __builtin_amdgcn_s_wait_tensorcnt((short)1);
    } else {
      wait_async_le0();
      if (wv0) __builtin_amdgcn_s_wait_tensorcnt((short)0);
    }
    __syncthreads();
#pragma unroll
    for (int kk = 0; kk < 64; kk += 32) {
      Frag a[4], bfr[2];
#pragma unroll
      for (int mi = 0; mi < 4; ++mi) {
        const bf16* p = &As[cur][wm * 64 + mi * 16 + l][kk + 8 * hh];
        a[mi].h[0] = *(const bf16x8*)p;         // K = kk + 8h .. +7
        a[mi].h[1] = *(const bf16x8*)(p + 16);  // K = kk + 16 + 8h .. +7
      }
#pragma unroll
      for (int ni = 0; ni < 2; ++ni) {
        const bf16* p = &Bs[cur][wn * 32 + ni * 16 + l][kk + 8 * hh];
        bfr[ni].h[0] = *(const bf16x8*)p;       // K = kk + 8h .. +7
        bfr[ni].h[1] = *(const bf16x8*)(p + 16);  // K = kk + 16 + 8h .. +7
      }
#pragma unroll
      for (int mi = 0; mi < 4; ++mi)
#pragma unroll
        for (int ni = 0; ni < 2; ++ni)
          acc[mi][ni] = wmma_bf16(a[mi].v, bfr[ni].v, acc[mi][ni]);
    }
    __syncthreads();  // all reads of buf `cur` done before it is restaged
  }
#pragma unroll
  for (int mi = 0; mi < 4; ++mi)
#pragma unroll
    for (int ni = 0; ni < 2; ++ni)
#pragma unroll
      for (int i = 0; i < 8; ++i) {
        int row = m0 + wm * 64 + mi * 16 + i + 8 * hh;
        int col = n0 + wn * 32 + ni * 16 + l;
        float v = acc[mi][ni][i];
        if (resid) v += resid[(size_t)row * N + col];
        Dm[(size_t)row * N + col] = v;
      }
}

// ---------------------------------------------------------------------------
// 6) RoPE (+ optional 1/sqrt(HD) scale) fp32 -> bf16. grid (T, nheads), 128 thr
// ---------------------------------------------------------------------------
__global__ __launch_bounds__(128)
void rope_kernel(const float* __restrict__ in, bf16* __restrict__ out,
                 const int* __restrict__ pos, int nh, float scale) {
  const int t = blockIdx.x, h = blockIdx.y, d = threadIdx.x;
  const float* x = in + ((size_t)t * nh + h) * HDIM;
  const float p = (float)pos[t];
  const int d0 = d & 63;
  float fr = __powf(10000.0f, -((float)(2 * d0)) / 128.0f);
  float sn, cs;
  __sincosf(p * fr, &sn, &cs);
  float v = x[d];
  float other = (d < 64) ? -x[d + 64] : x[d - 64];
  out[((size_t)t * nh + h) * HDIM + d] = (bf16)((v * cs + other * sn) * scale);
}

// V fp32 [T, KV*HD] -> bf16 transposed vT[kv][hd][t]
__global__ void vconv_kernel(const float* __restrict__ vf, bf16* __restrict__ vT) {
  size_t i = (size_t)blockIdx.x * 256 + threadIdx.x;
  if (i >= (size_t)KVH * HDIM * TT) return;
  size_t t = i % TT, rest = i / TT;  // rest = kv*HD + hd
  vT[i] = (bf16)vf[t * (KVH * HDIM) + rest];
}

// ---------------------------------------------------------------------------
// 7) Flash attention (causal, GQA). One wave = 16 query rows of one head.
//    QK^T and P*V via WMMA; P goes C-layout -> LDS -> A-layout (same wave).
// ---------------------------------------------------------------------------
__global__ __launch_bounds__(128)
void attn_kernel(const bf16* __restrict__ qb, const bf16* __restrict__ kb,
                 const bf16* __restrict__ vT, bf16* __restrict__ attn) {
  __shared__ __align__(16) bf16 Pbuf[4][16][40];  // row stride 80B
  const int w = threadIdx.x >> 5, lane = threadIdx.x & 31;
  const int l = lane & 15, hh = lane >> 4;
  const int head = blockIdx.y, kvh = head >> 2;
  const int qbase = (blockIdx.x * 4 + w) * 16;

  // Q A-fragments (pre-scaled by 1/sqrt(HD) in rope_kernel)
  Frag qa[4];
  const bf16* qrow = qb + (size_t)(qbase + l) * (HH * HDIM) + head * HDIM;
#pragma unroll
  for (int ks = 0; ks < 4; ++ks) {
    qa[ks].h[0] = *(const bf16x8*)(qrow + ks * 32 + 8 * hh);
    qa[ks].h[1] = *(const bf16x8*)(qrow + ks * 32 + 16 + 8 * hh);
  }

  float mrow[8], lrow[8];
  f32x8 acc[8];
#pragma unroll
  for (int i = 0; i < 8; ++i) { mrow[i] = -1e30f; lrow[i] = 0.f; }
#pragma unroll
  for (int j = 0; j < 8; ++j) acc[j] = fzero8();

  const int nkb = (qbase + 47) >> 5;  // key blocks of 32, covers keys <= qbase+15
  for (int kbk = 0; kbk < nkb; ++kbk) {
    const int key0 = kbk * 32;
    f32x8 s[2];
    s[0] = fzero8(); s[1] = fzero8();
#pragma unroll
    for (int nt = 0; nt < 2; ++nt) {
      const bf16* krow =
          kb + (size_t)(key0 + nt * 16 + l) * (KVH * HDIM) + kvh * HDIM + 8 * hh;
#pragma unroll
      for (int ks = 0; ks < 4; ++ks) {
        Frag kf;
        kf.h[0] = *(const bf16x8*)(krow + ks * 32);
        kf.h[1] = *(const bf16x8*)(krow + ks * 32 + 16);
        s[nt] = wmma_bf16(qa[ks].v, kf.v, s[nt]);
      }
    }
    // causal mask + online softmax (rows i+8h, cols across 16 lanes)
#pragma unroll
    for (int i = 0; i < 8; ++i) {
      int row = qbase + i + 8 * hh;
      float s0 = s[0][i]; if (key0 + l > row)      s0 = -1e30f;
      float s1 = s[1][i]; if (key0 + 16 + l > row) s1 = -1e30f;
      float r = fmaxf(s0, s1);
      r = fmaxf(r, __shfl_xor(r, 1));
      r = fmaxf(r, __shfl_xor(r, 2));
      r = fmaxf(r, __shfl_xor(r, 4));
      r = fmaxf(r, __shfl_xor(r, 8));
      float mnew  = fmaxf(mrow[i], r);
      float alpha = __expf(mrow[i] - mnew);
      float p0 = __expf(s0 - mnew), p1 = __expf(s1 - mnew);
      float rs = p0 + p1;
      rs += __shfl_xor(rs, 1);
      rs += __shfl_xor(rs, 2);
      rs += __shfl_xor(rs, 4);
      rs += __shfl_xor(rs, 8);
      lrow[i] = lrow[i] * alpha + rs;
      mrow[i] = mnew;
#pragma unroll
      for (int j = 0; j < 8; ++j) acc[j][i] *= alpha;
      Pbuf[w][i + 8 * hh][l]      = (bf16)p0;   // C layout -> LDS
      Pbuf[w][i + 8 * hh][16 + l] = (bf16)p1;
    }
    // A-layout read-back of P (same wave; compiler inserts dscnt wait)
    Frag pa;
    pa.h[0] = *(const bf16x8*)&Pbuf[w][l][8 * hh];
    pa.h[1] = *(const bf16x8*)&Pbuf[w][l][16 + 8 * hh];
    const bf16* vbase = vT + (size_t)kvh * HDIM * TT + key0 + 8 * hh;
#pragma unroll
    for (int j = 0; j < 8; ++j) {
      const bf16* vrow = vbase + (size_t)(j * 16 + l) * TT;
      Frag vf;
      vf.h[0] = *(const bf16x8*)vrow;
      vf.h[1] = *(const bf16x8*)(vrow + 16);
      acc[j] = wmma_bf16(pa.v, vf.v, acc[j]);
    }
  }
#pragma unroll
  for (int i = 0; i < 8; ++i) {
    float inv = 1.0f / lrow[i];
    int row = qbase + i + 8 * hh;
#pragma unroll
    for (int j = 0; j < 8; ++j)
      attn[(size_t)row * (HH * HDIM) + head * HDIM + j * 16 + l] =
          (bf16)(acc[j][i] * inv);
  }
}

// ---------------------------------------------------------------------------
// 8) SwiGLU combine, copy, scatter
// ---------------------------------------------------------------------------
__global__ void silu_kernel(const float* __restrict__ g, const float* __restrict__ u,
                            bf16* __restrict__ m) {
  size_t i = (size_t)blockIdx.x * 256 + threadIdx.x;
  if (i >= (size_t)TT * DFF) return;
  float gv = g[i];
  m[i] = (bf16)((gv / (1.0f + __expf(-gv))) * u[i]);
}

__global__ void copy_kernel(const float* __restrict__ src, float* __restrict__ dst,
                            size_t n) {
  size_t i = (size_t)blockIdx.x * 256 + threadIdx.x;
  if (i < n) dst[i] = src[i];
}

__global__ __launch_bounds__(256)
void scatter_kernel(const int* __restrict__ idxb, const float* __restrict__ gate,
                    const float* __restrict__ y, const float* __restrict__ sel,
                    float* __restrict__ out) {
  const int t = blockIdx.x, b = t / KSEL;
  const int s = idxb[t];
  const float g = gate[t];
  float* o = out + ((size_t)b * SS + s) * DD;
  const float* yp = y + (size_t)t * DD;
  const float* sp = sel + (size_t)t * DD;
  for (int d = threadIdx.x; d < DD; d += 256)
    o[d] += (yp[d] - sp[d]) * g;
}

// ---------------------------------------------------------------------------
// Host orchestration
// ---------------------------------------------------------------------------
extern "C" void kernel_launch(void* const* d_in, const int* in_sizes, int n_in,
                              void* d_out, int out_size, void* d_ws, size_t ws_size,
                              hipStream_t stream) {
  (void)in_sizes; (void)n_in; (void)out_size; (void)ws_size;
  const float* hidden   = (const float*)d_in[0];
  const float* original = (const float*)d_in[1];
  const float* posterior= (const float*)d_in[2];
  const float* prior    = (const float*)d_in[3];
  const int*   posids   = (const int*)d_in[4];
  const float* w_router = (const float*)d_in[5];
  const float* ln1      = (const float*)d_in[6];
  const float* ln2      = (const float*)d_in[7];
  const float* wq       = (const float*)d_in[8];
  const float* wk       = (const float*)d_in[9];
  const float* wv       = (const float*)d_in[10];
  const float* wo       = (const float*)d_in[11];
  const float* wg       = (const float*)d_in[12];
  const float* wu       = (const float*)d_in[13];
  const float* wd       = (const float*)d_in[14];
  float* out = (float*)d_out;

  char* p = (char*)d_ws;
  auto alloc = [&](size_t bytes) -> void* {
    void* r = (void*)p;
    p += (bytes + 255) & ~(size_t)255;
    return r;
  };
  float* scores = (float*)alloc((size_t)BB * SS * 4);
  int*   idxb   = (int*)alloc((size_t)TT * 4);
  float* gate   = (float*)alloc((size_t)TT * 4);
  int*   poss   = (int*)alloc((size_t)TT * 4);
  float* sel    = (float*)alloc((size_t)TT * DD * 4);
  bf16*  h1     = (bf16*)alloc((size_t)TT * DD * 2);
  float* x1     = (float*)alloc((size_t)TT * DD * 4);
  bf16*  h2     = (bf16*)alloc((size_t)TT * DD * 2);
  float* qf     = (float*)alloc((size_t)TT * HH * HDIM * 4);
  float* kf     = (float*)alloc((size_t)TT * KVH * HDIM * 4);
  float* vf     = (float*)alloc((size_t)TT * KVH * HDIM * 4);
  bf16*  qb     = (bf16*)alloc((size_t)TT * HH * HDIM * 2);
  bf16*  kbf    = (bf16*)alloc((size_t)TT * KVH * HDIM * 2);
  bf16*  vTb    = (bf16*)alloc((size_t)KVH * HDIM * TT * 2);
  bf16*  attn   = (bf16*)alloc((size_t)TT * HH * HDIM * 2);
  float* gbuf   = (float*)alloc((size_t)TT * DFF * 4);
  float* ubuf   = (float*)alloc((size_t)TT * DFF * 4);
  bf16*  mb     = (bf16*)alloc((size_t)TT * DFF * 2);
  float* ybuf   = (float*)alloc((size_t)TT * DD * 4);
  bf16*  wqT    = (bf16*)alloc((size_t)DD * (HH * HDIM) * 2);
  bf16*  wkT    = (bf16*)alloc((size_t)DD * (KVH * HDIM) * 2);
  bf16*  wvT    = (bf16*)alloc((size_t)DD * (KVH * HDIM) * 2);
  bf16*  woT    = (bf16*)alloc((size_t)(HH * HDIM) * DD * 2);
  bf16*  wgT    = (bf16*)alloc((size_t)DD * DFF * 2);
  bf16*  wuT    = (bf16*)alloc((size_t)DD * DFF * 2);
  bf16*  wdT    = (bf16*)alloc((size_t)DFF * DD * 2);

  auto cdiv = [](size_t a, size_t b) { return (unsigned)((a + b - 1) / b); };

  // router + selection
  router_kernel<<<BB * SS, 256, 0, stream>>>(original, posterior, prior, w_router,
                                             scores);
  topk_kernel<<<BB, 1024, 0, stream>>>(scores, idxb, gate);
  gather_rms_kernel<<<TT, 256, 0, stream>>>(hidden, idxb, posids, ln1, sel, h1,
                                            poss);

  // weight convert+transpose (fp32 [K,N] -> bf16 [N,K])
  wconvT_kernel<<<cdiv((size_t)DD * HH * HDIM, 256), 256, 0, stream>>>(wq, wqT, DD, HH * HDIM);
  wconvT_kernel<<<cdiv((size_t)DD * KVH * HDIM, 256), 256, 0, stream>>>(wk, wkT, DD, KVH * HDIM);
  wconvT_kernel<<<cdiv((size_t)DD * KVH * HDIM, 256), 256, 0, stream>>>(wv, wvT, DD, KVH * HDIM);
  wconvT_kernel<<<cdiv((size_t)HH * HDIM * DD, 256), 256, 0, stream>>>(wo, woT, HH * HDIM, DD);
  wconvT_kernel<<<cdiv((size_t)DD * DFF, 256), 256, 0, stream>>>(wg, wgT, DD, DFF);
  wconvT_kernel<<<cdiv((size_t)DD * DFF, 256), 256, 0, stream>>>(wu, wuT, DD, DFF);
  wconvT_kernel<<<cdiv((size_t)DFF * DD, 256), 256, 0, stream>>>(wd, wdT, DFF, DD);

  // QKV projections
  gemm_bf16_kernel<<<dim3((HH * HDIM) / 128, TT / 128), 256, 0, stream>>>(
      h1, wqT, qf, nullptr, TT, HH * HDIM, DD);
  gemm_bf16_kernel<<<dim3((KVH * HDIM) / 128, TT / 128), 256, 0, stream>>>(
      h1, wkT, kf, nullptr, TT, KVH * HDIM, DD);
  gemm_bf16_kernel<<<dim3((KVH * HDIM) / 128, TT / 128), 256, 0, stream>>>(
      h1, wvT, vf, nullptr, TT, KVH * HDIM, DD);

  // RoPE (q pre-scaled by 1/sqrt(HD)) and V transpose, all -> bf16
  rope_kernel<<<dim3(TT, HH), 128, 0, stream>>>(qf, qb, poss, HH,
                                                0.08838834764831845f);
  rope_kernel<<<dim3(TT, KVH), 128, 0, stream>>>(kf, kbf, poss, KVH, 1.0f);
  vconv_kernel<<<cdiv((size_t)KVH * HDIM * TT, 256), 256, 0, stream>>>(vf, vTb);

  // attention
  attn_kernel<<<dim3(TT / 64, HH), 128, 0, stream>>>(qb, kbf, vTb, attn);

  // output projection + residual
  gemm_bf16_kernel<<<dim3(DD / 128, TT / 128), 256, 0, stream>>>(
      attn, woT, x1, sel, TT, DD, HH * HDIM);

  // MLP
  rms2_kernel<<<TT, 256, 0, stream>>>(x1, ln2, h2);
  gemm_bf16_kernel<<<dim3(DFF / 128, TT / 128), 256, 0, stream>>>(
      h2, wgT, gbuf, nullptr, TT, DFF, DD);
  gemm_bf16_kernel<<<dim3(DFF / 128, TT / 128), 256, 0, stream>>>(
      h2, wuT, ubuf, nullptr, TT, DFF, DD);
  silu_kernel<<<cdiv((size_t)TT * DFF, 256), 256, 0, stream>>>(gbuf, ubuf, mb);
  gemm_bf16_kernel<<<dim3(DD / 128, TT / 128), 256, 0, stream>>>(
      mb, wdT, ybuf, x1, TT, DD, DFF);

  // write back: out = hidden; out[sel] += (y - sel) * gate
  copy_kernel<<<cdiv((size_t)BB * SS * DD, 256), 256, 0, stream>>>(
      hidden, out, (size_t)BB * SS * DD);
  scatter_kernel<<<TT, 256, 0, stream>>>(idxb, gate, ybuf, sel, out);
}
